// MultiHeadAttention_31988916421188
// MI455X (gfx1250) — matrix operand order repaired and checked by reading, not compiled, not run on hardware
//
#include <hip/hip_runtime.h>

// Problem constants (match reference)
#define BATCH 4
#define SEQ   2048
#define DIM   1024

typedef __bf16 bf16_t;
typedef __attribute__((ext_vector_type(8)))  __bf16 v8bf;
typedef __attribute__((ext_vector_type(16))) __bf16 v16bf;
typedef __attribute__((ext_vector_type(8)))  float  v8f;

__device__ __forceinline__ unsigned short f32_to_bf16_bits(float f) {
    unsigned u = __float_as_uint(f);
    u += 0x7FFFu + ((u >> 16) & 1u);       // round-to-nearest-even
    return (unsigned short)(u >> 16);
}
__device__ __forceinline__ float bf16_bits_to_f32(unsigned short h) {
    return __uint_as_float(((unsigned)h) << 16);
}

// ---------------------------------------------------------------------------
// WMMA fragment loaders (contiguous-K order shared by A and B, wave32)
// A 16x32 bf16: lane L -> row (L&15); contiguous K = (L>>4)*16 .. +15
__device__ __forceinline__ v16bf load_frag_a(const bf16_t* __restrict__ A,
                                             int lda, int m0, int k0, int lane) {
    const bf16_t* p = A + (size_t)(m0 + (lane & 15)) * lda + (k0 + ((lane >> 4) << 4));
    v8bf lo = *(const v8bf*)p;
    v8bf hi = *(const v8bf*)(p + 8);
    return __builtin_shufflevector(lo, hi, 0,1,2,3,4,5,6,7,8,9,10,11,12,13,14,15);
}
// B 32x16 bf16 read from a K-contiguous (transposed / "Bt") matrix:
// lane L -> column (L&15); holds contiguous K = (L>>4)*16 .. +15 (32B aligned)
__device__ __forceinline__ v16bf load_frag_b(const bf16_t* __restrict__ Bt,
                                             int ldb, int n0, int k0, int lane) {
    const bf16_t* p = Bt + (size_t)(n0 + (lane & 15)) * ldb + (k0 + ((lane >> 4) << 4));
    return *(const v16bf*)p;
}

// ---------------------------------------------------------------------------
// Generic batched bf16 WMMA GEMM: C = A(MxK, row-major) * Bt(NxK, row-major)^T
// Block = 128 threads (4 waves); wave tile 64x64; block tile 128x128.
// MODE: 0 = store bf16          (C: ushort, row-major, ldc)
//       1 = store bf16 * scale  (scores)
//       2 = store bf16 transposed: C[n*ldc + m]  (builds Vt)
//       3 = store f32 + bias[n] (final projection)
template<int MODE>
__global__ __launch_bounds__(128, 1)
void gemm_bf16_wmma(const bf16_t* __restrict__ A, const bf16_t* __restrict__ Bt,
                    void* __restrict__ C,
                    int lda, int ldb, int ldc, int K,
                    long aBatch, long bBatch, long cBatch,
                    float scale, const float* __restrict__ bias)
{
    const int lane = threadIdx.x & 31;
    const int wave = threadIdx.x >> 5;
    const int z    = blockIdx.z;
    A  += (size_t)z * aBatch;
    Bt += (size_t)z * bBatch;

    const int m0 = blockIdx.y * 128 + (wave >> 1) * 64;
    const int n0 = blockIdx.x * 128 + (wave & 1) * 64;

    v8f c[4][4];
    #pragma unroll
    for (int i = 0; i < 4; ++i)
        #pragma unroll
        for (int j = 0; j < 4; ++j) {
            v8f zv = {0.f,0.f,0.f,0.f,0.f,0.f,0.f,0.f};
            c[i][j] = zv;
        }

    for (int k0 = 0; k0 < K; k0 += 32) {
        v16bf a[4], b[4];
        #pragma unroll
        for (int i = 0; i < 4; ++i) a[i] = load_frag_a(A,  lda, m0 + i*16, k0, lane);
        #pragma unroll
        for (int j = 0; j < 4; ++j) b[j] = load_frag_b(Bt, ldb, n0 + j*16, k0, lane);
        #pragma unroll
        for (int i = 0; i < 4; ++i)
            #pragma unroll
            for (int j = 0; j < 4; ++j)
                c[i][j] = __builtin_amdgcn_wmma_f32_16x16x32_bf16(
                    false, a[i], false, b[j], (short)0, c[i][j], false, false);
    }

    // C/D layout: VGPR r -> row r + 8*(lane>=16); col = lane&15
    const int rbase = (lane >> 4) << 3;
    const int ncol  = lane & 15;
    #pragma unroll
    for (int i = 0; i < 4; ++i) {
        #pragma unroll
        for (int j = 0; j < 4; ++j) {
            const int mrow0 = m0 + i*16 + rbase;
            const int ncc   = n0 + j*16 + ncol;
            #pragma unroll
            for (int r = 0; r < 8; ++r) {
                const float  val = c[i][j][r];
                const size_t row = (size_t)(mrow0 + r);
                if (MODE == 0) {
                    ((unsigned short*)C)[(size_t)cBatch * z + row * ldc + ncc] =
                        f32_to_bf16_bits(val);
                } else if (MODE == 1) {
                    ((unsigned short*)C)[(size_t)cBatch * z + row * ldc + ncc] =
                        f32_to_bf16_bits(val * scale);
                } else if (MODE == 2) {
                    ((unsigned short*)C)[(size_t)cBatch * z + (size_t)ncc * ldc + row] =
                        f32_to_bf16_bits(val);
                } else {
                    ((float*)C)[(size_t)cBatch * z + row * ldc + ncc] = val + bias[ncc];
                }
            }
        }
    }
}

// ---------------------------------------------------------------------------
__global__ void convert_f32_to_bf16(const float* __restrict__ in,
                                    unsigned short* __restrict__ out, int n) {
    int i = blockIdx.x * blockDim.x + threadIdx.x;
    if (i < n) out[i] = f32_to_bf16_bits(in[i]);
}

// Wt[n*d + k] = bf16(W[k*d + n])  (coalesced reads over n)
__global__ void transpose_convert(const float* __restrict__ W,
                                  unsigned short* __restrict__ Wt, int d) {
    int idx = blockIdx.x * blockDim.x + threadIdx.x;
    if (idx < d * d) {
        int k = idx / d, n = idx % d;
        Wt[(size_t)n * d + k] = f32_to_bf16_bits(W[idx]);
    }
}

// In-place row softmax on bf16 scores; row length fixed = SEQ, block = 256 (8 waves).
__global__ __launch_bounds__(256)
void softmax_rows_bf16(unsigned short* __restrict__ S) {
    const int   tid = threadIdx.x;
    const int   wv  = tid >> 5, ln = tid & 31;
    unsigned short* p = S + (size_t)blockIdx.x * SEQ;

    float v[8];
    float m = -3.402823466e38f;
    #pragma unroll
    for (int i = 0; i < 8; ++i) {
        v[i] = bf16_bits_to_f32(p[tid + i * 256]);
        m = fmaxf(m, v[i]);
    }
    __shared__ float red[8];
    #pragma unroll
    for (int off = 16; off > 0; off >>= 1) m = fmaxf(m, __shfl_xor(m, off, 32));
    if (ln == 0) red[wv] = m;
    __syncthreads();
    m = red[0];
    #pragma unroll
    for (int i = 1; i < 8; ++i) m = fmaxf(m, red[i]);
    __syncthreads();                       // before reusing red[]

    float s = 0.f;
    #pragma unroll
    for (int i = 0; i < 8; ++i) { v[i] = __expf(v[i] - m); s += v[i]; }
    #pragma unroll
    for (int off = 16; off > 0; off >>= 1) s += __shfl_xor(s, off, 32);
    if (ln == 0) red[wv] = s;
    __syncthreads();
    float tot = red[0];
    #pragma unroll
    for (int i = 1; i < 8; ++i) tot += red[i];
    const float inv = 1.0f / tot;
    #pragma unroll
    for (int i = 0; i < 8; ++i) p[tid + i * 256] = f32_to_bf16_bits(v[i] * inv);
}

// ---------------------------------------------------------------------------
extern "C" void kernel_launch(void* const* d_in, const int* in_sizes, int n_in,
                              void* d_out, int out_size, void* d_ws, size_t ws_size,
                              hipStream_t stream)
{
    const float* X  = (const float*)d_in[0];
    const float* Wq = (const float*)d_in[1];
    const float* Wk = (const float*)d_in[2];
    const float* Wv = (const float*)d_in[3];
    const float* Wo = (const float*)d_in[4];
    const float* bo = (const float*)d_in[5];

    const size_t BSD = (size_t)BATCH * SEQ * DIM;   // 8,388,608
    const size_t DD  = (size_t)DIM * DIM;           // 1,048,576
    const size_t BSS = (size_t)BATCH * SEQ * SEQ;   // 16,777,216

    char*  ws  = (char*)d_ws;
    size_t off = 0;
    auto alloc = [&](size_t bytes) -> void* {
        void* p = ws + off;
        off += (bytes + 255) & ~(size_t)255;
        return p;
    };
    unsigned short* Xb  = (unsigned short*)alloc(BSD * 2);  // X  bf16
    unsigned short* Wqt = (unsigned short*)alloc(DD  * 2);  // Wq^T bf16
    unsigned short* Wkt = (unsigned short*)alloc(DD  * 2);
    unsigned short* Wvt = (unsigned short*)alloc(DD  * 2);
    unsigned short* Wot = (unsigned short*)alloc(DD  * 2);
    unsigned short* Qb  = (unsigned short*)alloc(BSD * 2);  // Q  [b][s][d]
    unsigned short* Kb  = (unsigned short*)alloc(BSD * 2);  // K  [b][s][d]
    unsigned short* Vt  = (unsigned short*)alloc(BSD * 2);  // V^T [b][d][s]
    unsigned short* Ob  = (unsigned short*)alloc(BSD * 2);  // attn out bf16
    unsigned short* Sc  = (unsigned short*)alloc(BSS * 2);  // scores / probs bf16

    // 1) precision conversion + weight transposes
    convert_f32_to_bf16<<<(unsigned)((BSD + 255) / 256), 256, 0, stream>>>(X, Xb, (int)BSD);
    transpose_convert  <<<(unsigned)((DD  + 255) / 256), 256, 0, stream>>>(Wq, Wqt, DIM);
    transpose_convert  <<<(unsigned)((DD  + 255) / 256), 256, 0, stream>>>(Wk, Wkt, DIM);
    transpose_convert  <<<(unsigned)((DD  + 255) / 256), 256, 0, stream>>>(Wv, Wvt, DIM);
    transpose_convert  <<<(unsigned)((DD  + 255) / 256), 256, 0, stream>>>(Wo, Wot, DIM);

    const dim3 blk(128);
    const long SD = (long)SEQ * DIM;
    const long SS = (long)SEQ * SEQ;

    // 2) Q = X*Wq, K = X*Wk (row-major), V^T (transposed store, batched)
    gemm_bf16_wmma<0><<<dim3(DIM/128, (BATCH*SEQ)/128, 1), blk, 0, stream>>>(
        (const bf16_t*)Xb, (const bf16_t*)Wqt, Qb, DIM, DIM, DIM, DIM,
        0, 0, 0, 1.0f, nullptr);
    gemm_bf16_wmma<0><<<dim3(DIM/128, (BATCH*SEQ)/128, 1), blk, 0, stream>>>(
        (const bf16_t*)Xb, (const bf16_t*)Wkt, Kb, DIM, DIM, DIM, DIM,
        0, 0, 0, 1.0f, nullptr);
    gemm_bf16_wmma<2><<<dim3(DIM/128, SEQ/128, BATCH), blk, 0, stream>>>(
        (const bf16_t*)Xb, (const bf16_t*)Wvt, Vt, DIM, DIM, SEQ, DIM,
        SD, 0, SD, 1.0f, nullptr);

    // 3) scores = (Q K^T) / sqrt(D): K row-major is exactly the Bt layout
    gemm_bf16_wmma<1><<<dim3(SEQ/128, SEQ/128, BATCH), blk, 0, stream>>>(
        (const bf16_t*)Qb, (const bf16_t*)Kb, Sc, DIM, DIM, SEQ, DIM,
        SD, SD, SS, 0.03125f /* 1/sqrt(1024) */, nullptr);

    // 4) softmax over each of the B*S rows, in place
    softmax_rows_bf16<<<BATCH * SEQ, 256, 0, stream>>>(Sc);

    // 5) O = P * V  (Vt supplies contiguous-K B fragments)
    gemm_bf16_wmma<0><<<dim3(DIM/128, SEQ/128, BATCH), blk, 0, stream>>>(
        (const bf16_t*)Sc, (const bf16_t*)Vt, Ob, SEQ, SEQ, DIM, SEQ,
        SS, SD, SD, 1.0f, nullptr);

    // 6) out = O * Wo + bo  (f32 output)
    gemm_bf16_wmma<3><<<dim3(DIM/128, (BATCH*SEQ)/128, 1), blk, 0, stream>>>(
        (const bf16_t*)Ob, (const bf16_t*)Wot, d_out, DIM, DIM, DIM, DIM,
        0, 0, 0, 1.0f, bo);
}
